// Attention_6897717477861
// MI455X (gfx1250) — matrix operand order repaired and checked by reading, not compiled, not run on hardware
//
#include <hip/hip_runtime.h>
#include <hip/hip_bf16.h>

typedef __bf16 bf16;
typedef __attribute__((ext_vector_type(16))) __bf16 v16bf;
typedef __attribute__((ext_vector_type(8)))  __bf16 v8bf;
typedef __attribute__((ext_vector_type(8)))  float  v8f;
typedef __attribute__((ext_vector_type(4)))  unsigned int v4u;
typedef __attribute__((ext_vector_type(8)))  int v8i;
typedef __attribute__((ext_vector_type(4)))  int v4i;

constexpr int B_ = 8, S_ = 4096, E_ = 128;
constexpr int ROWS = B_ * S_;

// LDS row strides (elements) with +16B padding to rotate banks row-to-row.
constexpr int KROW = 128 + 8;  // 272 B  (TDM pad: every 64 DW insert 4 DW)
constexpr int VROW = 64 + 8;   // 144 B  (TDM pad: every 32 DW insert 4 DW)
constexpr int PROW = 64 + 8;   // manual padding for P staging buffer

__device__ inline v8f vzero8() {
  v8f z;
#pragma unroll
  for (int i = 0; i < 8; ++i) z[i] = 0.0f;
  return z;
}

// ---------------------------------------------------------------------------
// Tensor Data Mover: 2D tile load Global -> LDS (D# per CDNA5 ISA ch.8).
//   group0: [count=1][lds_addr][global_addr 57b][type=2]
//   group1: data_size=2B, pad_enable, pad_interval/amount, dims/strides
//   groups 2/3 (+extra group on this toolchain): zero (2D tensor)
// ---------------------------------------------------------------------------
__device__ inline void tdm_load_2d(unsigned int lds_byte_off, const void* gptr,
                                   int td0, int td1, long long stride0_elems,
                                   int tile0, int tile1,
                                   int pad_interval_code, int pad_amount_code)
{
  unsigned long long ga = (unsigned long long)(uintptr_t)gptr;
  v4u g0;
  g0[0] = 1u;                                   // count=1 (valid user D#)
  g0[1] = lds_byte_off;                         // LDS dest (bytes)
  g0[2] = (unsigned int)ga;                     // global addr [31:0]
  g0[3] = (unsigned int)((ga >> 32) & 0x01FFFFFFu) | (2u << 30);  // [56:32]|type=2

  unsigned long long s0 = (unsigned long long)stride0_elems;
  v8i g1;
  g1[0] = (int)((1u << 16)                      // data_size = 2 bytes
              | (1u << 20)                      // pad_enable
              | ((unsigned)pad_interval_code << 22)
              | ((unsigned)pad_amount_code << 25));
  g1[1] = (int)(((unsigned)td0 & 0xFFFFu) << 16);                       // td0 lo
  g1[2] = (int)((((unsigned)td0 >> 16) & 0xFFFFu)
              | (((unsigned)td1 & 0xFFFFu) << 16));                     // td0 hi | td1 lo
  g1[3] = (int)((((unsigned)td1 >> 16) & 0xFFFFu)
              | (((unsigned)tile0 & 0xFFFFu) << 16));                   // td1 hi | tile0
  g1[4] = (int)((unsigned)tile1 & 0xFFFFu);                             // tile1 | tile2=0
  g1[5] = (int)(unsigned int)s0;                                        // dim0 stride lo
  g1[6] = (int)((unsigned int)(s0 >> 32) & 0xFFFFu);                    // stride hi | td1s=0
  g1[7] = 0;
  v4i z4 = {0, 0, 0, 0};
  v8i z8 = {0, 0, 0, 0, 0, 0, 0, 0};
  __builtin_amdgcn_tensor_load_to_lds(g0, g1, z4, z4, z8, 0);
}

// ---------------------------------------------------------------------------
// Kernel 1: fused Q/K/V projection: y = PReLU(x @ W^T + b), fp32 -> bf16.
// qx, kx stored [B*S][E]; vx stored transposed [B][E][S].
// ---------------------------------------------------------------------------
__global__ void __launch_bounds__(256)
proj_kernel(const float* __restrict__ x,
            const float* __restrict__ Wq, const float* __restrict__ bq, const float* __restrict__ aq,
            const float* __restrict__ Wk, const float* __restrict__ bk, const float* __restrict__ ak,
            const float* __restrict__ Wv, const float* __restrict__ bv, const float* __restrict__ av,
            bf16* __restrict__ qx, bf16* __restrict__ kx, bf16* __restrict__ vT)
{
  const int lane = threadIdx.x & 31;
  const int wave = threadIdx.x >> 5;
  const int row0 = blockIdx.x * 128 + wave * 16;
  const int M    = lane & 15;
  const int half = lane >> 4;

  v16bf afrag[4];
#pragma unroll
  for (int ks = 0; ks < 4; ++ks) {
    const float* plo = x + (size_t)(row0 + M) * E_ + ks * 32 + half * 8;
    const float* phi = plo + 16;
    v16bf a;
#pragma unroll
    for (int i = 0; i < 8; ++i) { a[i] = (bf16)plo[i]; a[8 + i] = (bf16)phi[i]; }
    afrag[ks] = a;
  }

  const float* Ws[3] = {Wq, Wk, Wv};
  const float* bs[3] = {bq, bk, bv};
  const float* as[3] = {aq, ak, av};

#pragma unroll
  for (int p = 0; p < 3; ++p) {
    const float* W = Ws[p];
    v8f acc[8];
#pragma unroll
    for (int nt = 0; nt < 8; ++nt) acc[nt] = vzero8();

#pragma unroll
    for (int ks = 0; ks < 4; ++ks) {
#pragma unroll
      for (int nt = 0; nt < 8; ++nt) {
        const float* wp = W + (size_t)(nt * 16 + M) * E_ + ks * 32 + half * 8;
        v16bf bfrag;
#pragma unroll
        for (int i = 0; i < 16; ++i) bfrag[i] = (bf16)wp[i < 8 ? i : i + 8];
        acc[nt] = __builtin_amdgcn_wmma_f32_16x16x32_bf16(
            false, afrag[ks], false, bfrag, (short)0, acc[nt], false, false);
      }
    }

    const float slope = *as[p];
#pragma unroll
    for (int nt = 0; nt < 8; ++nt) {
      const int col = nt * 16 + M;
      const float bias = bs[p][col];
#pragma unroll
      for (int r = 0; r < 8; ++r) {
        float y = acc[nt][r] + bias;
        y = (y >= 0.0f) ? y : slope * y;
        const int gr = row0 + r + 8 * half;
        if (p == 2) {
          const int bb = gr >> 12;
          const int ss = gr & (S_ - 1);
          vT[((size_t)bb * E_ + col) * S_ + ss] = (bf16)y;
        } else if (p == 0) {
          qx[(size_t)gr * E_ + col] = (bf16)y;
        } else {
          kx[(size_t)gr * E_ + col] = (bf16)y;
        }
      }
    }
  }
}

// ---------------------------------------------------------------------------
// Kernel 2: flash attention with TDM double-buffered K/V staging in LDS.
// grid = (S/128, B), 256 threads = 8 waves; each wave owns 16 query rows.
// Wave 0 drives the Tensor Data Mover; tiles published via __syncthreads().
// ---------------------------------------------------------------------------
__global__ void __launch_bounds__(256)
attn_kernel(const bf16* __restrict__ qx, const bf16* __restrict__ kx,
            const bf16* __restrict__ vT, float* __restrict__ out)
{
  __shared__ __align__(32) struct {
    bf16 k[2][64 * KROW];    // key tile:  64 keys x 128 e (padded rows)
    bf16 v[2][128 * VROW];   // value tile: 128 e x 64 keys (padded rows)
    bf16 p[8][16 * PROW];    // per-wave P staging (padded rows)
  } sm;

  const int lane = threadIdx.x & 31;
  const int wave = threadIdx.x >> 5;
  const int b    = blockIdx.y;
  const int q0   = blockIdx.x * 128 + wave * 16;
  const int M    = lane & 15;
  const int half = lane >> 4;

  const bf16* qb = qx + (size_t)b * S_ * E_;
  const bf16* kb = kx + (size_t)b * S_ * E_;
  const bf16* vb = vT + (size_t)b * E_ * S_;

  // Preload Q A-fragments once.
  v16bf qfrag[4];
#pragma unroll
  for (int ks = 0; ks < 4; ++ks) {
    const bf16* plo = qb + (size_t)(q0 + M) * E_ + ks * 32 + half * 8;
    v8bf lo = *(const v8bf*)plo;
    v8bf hi = *(const v8bf*)(plo + 16);
    v16bf a;
#pragma unroll
    for (int i = 0; i < 8; ++i) { a[i] = lo[i]; a[8 + i] = hi[i]; }
    qfrag[ks] = a;
  }

  v8f oacc[8];
#pragma unroll
  for (int nt = 0; nt < 8; ++nt) oacc[nt] = vzero8();
  float mrun[8], lrun[8];
#pragma unroll
  for (int r = 0; r < 8; ++r) { mrun[r] = -3.0e38f; lrun[r] = 0.0f; }

  const float scale = 0.08838834764831845f;  // 1/sqrt(128)
  bf16* pb = &sm.p[wave][0];
  constexpr int NT = S_ / 64;

  // Prologue: TDM fills stage 0 (K: 128x64 tile, V^T: 64x128 tile).
  if (wave == 0) {
    tdm_load_2d((unsigned int)(uintptr_t)&sm.k[0][0], kb,
                E_, 64, E_, E_, 64, /*pad every 64DW*/5, /*+4DW*/3);
    tdm_load_2d((unsigned int)(uintptr_t)&sm.v[0][0], vb,
                64, E_, S_, 64, E_, /*pad every 32DW*/4, /*+4DW*/3);
  }

  for (int kt = 0; kt < NT; ++kt) {
    const int cur = kt & 1;

    if (wave == 0) __builtin_amdgcn_s_wait_tensorcnt(0);
    __syncthreads();  // stage `cur` ready for everyone; stage `cur^1` drained

    // Kick off next tile's DMA so it overlaps this tile's compute.
    if (wave == 0 && kt + 1 < NT) {
      const int kn = (kt + 1) * 64;
      tdm_load_2d((unsigned int)(uintptr_t)&sm.k[cur ^ 1][0],
                  kb + (size_t)kn * E_, E_, 64, E_, E_, 64, 5, 3);
      tdm_load_2d((unsigned int)(uintptr_t)&sm.v[cur ^ 1][0],
                  vb + kn, 64, E_, S_, 64, E_, 4, 3);
    }

    // ---- S = Q @ K^T  (16 x 64), K fragments from LDS ----
    v8f sacc[4];
#pragma unroll
    for (int nt = 0; nt < 4; ++nt) sacc[nt] = vzero8();
#pragma unroll
    for (int ks = 0; ks < 4; ++ks) {
#pragma unroll
      for (int nt = 0; nt < 4; ++nt) {
        const bf16* kp = &sm.k[cur][(nt * 16 + M) * KROW + ks * 32 + half * 8];
        v8bf lo = *(const v8bf*)kp;
        v8bf hi = *(const v8bf*)(kp + 16);
        v16bf kfrag;
#pragma unroll
        for (int i = 0; i < 8; ++i) { kfrag[i] = lo[i]; kfrag[8 + i] = hi[i]; }
        sacc[nt] = __builtin_amdgcn_wmma_f32_16x16x32_bf16(
            false, qfrag[ks], false, kfrag, (short)0, sacc[nt], false, false);
      }
    }

    // ---- online softmax ----
    float rmax[8];
#pragma unroll
    for (int r = 0; r < 8; ++r) {
      float m = -3.0e38f;
#pragma unroll
      for (int nt = 0; nt < 4; ++nt) {
        sacc[nt][r] *= scale;
        m = fmaxf(m, sacc[nt][r]);
      }
      rmax[r] = m;
    }
#pragma unroll
    for (int d = 1; d < 16; d <<= 1) {
#pragma unroll
      for (int r = 0; r < 8; ++r)
        rmax[r] = fmaxf(rmax[r], __shfl_xor(rmax[r], d, 32));
    }

    float rsum[8], corr[8];
#pragma unroll
    for (int r = 0; r < 8; ++r) {
      const float mnew = fmaxf(mrun[r], rmax[r]);
      corr[r] = __expf(mrun[r] - mnew);
      mrun[r] = mnew;
      float s = 0.0f;
#pragma unroll
      for (int nt = 0; nt < 4; ++nt) {
        const float e = __expf(sacc[nt][r] - mnew);
        sacc[nt][r] = e;
        s += e;
      }
      rsum[r] = s;
    }
#pragma unroll
    for (int d = 1; d < 16; d <<= 1) {
#pragma unroll
      for (int r = 0; r < 8; ++r)
        rsum[r] += __shfl_xor(rsum[r], d, 32);
    }
#pragma unroll
    for (int r = 0; r < 8; ++r) {
      lrun[r] = lrun[r] * corr[r] + rsum[r];
#pragma unroll
      for (int nt = 0; nt < 8; ++nt) oacc[nt][r] *= corr[r];
    }

    // ---- stage P (bf16) through per-wave LDS: C-layout -> row-major ----
#pragma unroll
    for (int nt = 0; nt < 4; ++nt)
#pragma unroll
      for (int r = 0; r < 8; ++r)
        pb[(r + 8 * half) * PROW + nt * 16 + M] = (bf16)sacc[nt][r];

    // ---- O += P @ V, V fragments from LDS (transposed layout) ----
#pragma unroll
    for (int ks2 = 0; ks2 < 2; ++ks2) {
      const bf16* plo = pb + M * PROW + ks2 * 32 + half * 8;
      v8bf lo = *(const v8bf*)plo;
      v8bf hi = *(const v8bf*)(plo + 16);
      v16bf pfrag;
#pragma unroll
      for (int i = 0; i < 8; ++i) { pfrag[i] = lo[i]; pfrag[8 + i] = hi[i]; }
#pragma unroll
      for (int nt = 0; nt < 8; ++nt) {
        const bf16* vp = &sm.v[cur][(nt * 16 + M) * VROW + ks2 * 32 + half * 8];
        v8bf vl = *(const v8bf*)vp;
        v8bf vh = *(const v8bf*)(vp + 16);
        v16bf vfrag;
#pragma unroll
        for (int i = 0; i < 8; ++i) { vfrag[i] = vl[i]; vfrag[8 + i] = vh[i]; }
        oacc[nt] = __builtin_amdgcn_wmma_f32_16x16x32_bf16(
            false, pfrag, false, vfrag, (short)0, oacc[nt], false, false);
      }
    }
  }

  // ---- epilogue: normalize and store fp32 ----
  float inv[8];
#pragma unroll
  for (int r = 0; r < 8; ++r) inv[r] = 1.0f / lrun[r];
#pragma unroll
  for (int nt = 0; nt < 8; ++nt) {
    const int col = nt * 16 + M;
#pragma unroll
    for (int r = 0; r < 8; ++r) {
      const int gq = q0 + r + 8 * half;
      out[((size_t)b * S_ + gq) * E_ + col] = oacc[nt][r] * inv[r];
    }
  }
}

extern "C" void kernel_launch(void* const* d_in, const int* in_sizes, int n_in,
                              void* d_out, int out_size, void* d_ws, size_t ws_size,
                              hipStream_t stream) {
  const float* x  = (const float*)d_in[0];
  const float* Wq = (const float*)d_in[1];
  const float* bq = (const float*)d_in[2];
  const float* aq = (const float*)d_in[3];
  const float* Wk = (const float*)d_in[4];
  const float* bk = (const float*)d_in[5];
  const float* ak = (const float*)d_in[6];
  const float* Wv = (const float*)d_in[7];
  const float* bv = (const float*)d_in[8];
  const float* av = (const float*)d_in[9];
  float* out = (float*)d_out;

  const size_t N = (size_t)B_ * S_ * E_;
  bf16* qx = (bf16*)d_ws;
  bf16* kx = qx + N;
  bf16* vT = kx + N;

  proj_kernel<<<ROWS / 128, 256, 0, stream>>>(x, Wq, bq, aq, Wk, bk, ak,
                                              Wv, bv, av, qx, kx, vT);
  dim3 grid2(S_ / 128, B_);
  attn_kernel<<<grid2, 256, 0, stream>>>(qx, kx, vT, out);
}
